// CHIVE_53111565583018
// MI455X (gfx1250) — matrix operand order repaired and checked by reading, not compiled, not run on hardware
//
#include <hip/hip_runtime.h>
#include <hip/hip_bf16.h>
#include <math.h>

// ---------------------------------------------------------------------------
// CDNA5 (gfx1250) implementation of the clockwork-RNN encoder/decoder.
//  K1: WMMA GEMM  - input projections (parallel over T)
//  K2: wave32 serial encoder scan (weights in LDS) + stream compaction
//  K3: WMMA GEMM  - decoder mean/logvar/z -> hs_new -> hp_new (parallel over K)
//  K4: wave32 serial decoder scan (hff/hfc/hp_d select) + KL of last row
// ---------------------------------------------------------------------------

typedef __attribute__((ext_vector_type(16))) _Float16 v16h;
typedef __attribute__((ext_vector_type(8)))  float    v8f;

#define H 32

__device__ __forceinline__ v8f wmma_f16(v16h a, v16h b, v8f c) {
  // (neg_a, A, neg_b, B, c_mod, C, reuse_a, reuse_b) -> v_wmma_f32_16x16x32_f16
  return __builtin_amdgcn_wmma_f32_16x16x32_f16(false, a, false, b, (short)0, c,
                                                false, false);
}

__device__ __forceinline__ float gelu_exact(float v) {
  return 0.5f * v * (1.0f + erff(v * 0.70710678118654752440f));
}

// A fragment (16x32 f16, M x K): lane L holds row m=L&15; K slices per ISA:
//   elems 0..7  -> K = e + 8*half ;  elems 8..15 -> K = 16 + (e-8) + 8*half
__device__ __forceinline__ v16h afrag_f32(const float* __restrict__ src, int ldr,
                                          int kmax, int mmax, int lane) {
  int m = lane & 15, half = lane >> 4;
  v16h a;
#pragma unroll
  for (int e = 0; e < 16; ++e) {
    int k = (e < 8) ? (e + 8 * half) : (e + 8 + 8 * half);
    float v = (m < mmax && k < kmax) ? src[m * ldr + k] : 0.0f;
    a[e] = (_Float16)v;
  }
  return a;
}

__device__ __forceinline__ v16h afrag_lds(const _Float16* __restrict__ src, int lane) {
  int m = lane & 15, half = lane >> 4;
  v16h a;
#pragma unroll
  for (int e = 0; e < 16; ++e) {
    int k = (e < 8) ? (e + 8 * half) : (e + 8 + 8 * half);
    a[e] = src[m * 32 + k];
  }
  return a;
}

// B fragment (32x16 f16, K x N): lane L holds col n=(L&15)+n0; K order same as A frag
__device__ __forceinline__ v16h bfrag_f32(const float* __restrict__ W, int ldn,
                                          int n0, int kmax, int lane) {
  int n = (lane & 15) + n0, half = lane >> 4;
  v16h b;
#pragma unroll
  for (int e = 0; e < 16; ++e) {
    int k = (e < 8) ? (e + 8 * half) : (e + 8 + 8 * half);
    float v = (k < kmax) ? W[k * ldn + n] : 0.0f;
    b[e] = (_Float16)v;
  }
  return b;
}

// ---------------------------------------------------------------------------
// K1: xfp = frnn_seq @ Wxf0 + bxf0 ; xpp = phrnn_seq @ Wxp0 + bxp0
// one wave per 16-row tile; 4 WMMAs per wave (2 outputs x 2 N-tiles)
// ---------------------------------------------------------------------------
__global__ void k1_inproj(const float* __restrict__ frnn_seq,
                          const float* __restrict__ phrnn_seq,
                          const float* __restrict__ Wxf0, const float* __restrict__ bxf0,
                          const float* __restrict__ Wxp0, const float* __restrict__ bxp0,
                          float* __restrict__ xfp, float* __restrict__ xpp, int T) {
  int lane = threadIdx.x & 31;
  int wave = blockIdx.x * (blockDim.x >> 5) + (threadIdx.x >> 5);
  int row0 = wave * 16;
  if (row0 >= T) return;
  int mmax = T - row0; if (mmax > 16) mmax = 16;

  v16h Bf0 = bfrag_f32(Wxf0, 32, 0, 13, lane);
  v16h Bf1 = bfrag_f32(Wxf0, 32, 16, 13, lane);
  v16h Bp0 = bfrag_f32(Wxp0, 32, 0, 7, lane);
  v16h Bp1 = bfrag_f32(Wxp0, 32, 16, 7, lane);

  v16h Af = afrag_f32(frnn_seq + (long)row0 * 13, 13, 13, mmax, lane);
  v16h Ap = afrag_f32(phrnn_seq + (long)row0 * 7, 7, 7, mmax, lane);

  v8f c0 = {}, c1 = {}, c2 = {}, c3 = {};
  c0 = wmma_f16(Af, Bf0, c0);
  c1 = wmma_f16(Af, Bf1, c1);
  c2 = wmma_f16(Ap, Bp0, c2);
  c3 = wmma_f16(Ap, Bp1, c3);

  int half = lane >> 4, n = lane & 15;
#pragma unroll
  for (int r = 0; r < 8; ++r) {
    int lrow = r + 8 * half;
    if (lrow >= mmax) continue;
    long row = row0 + lrow;
    xfp[row * H + n]      = c0[r] + bxf0[n];
    xfp[row * H + 16 + n] = c1[r] + bxf0[16 + n];
    xpp[row * H + n]      = c2[r] + bxp0[n];
    xpp[row * H + 16 + n] = c3[r] + bxp0[16 + n];
  }
}

// ---------------------------------------------------------------------------
// K2: serial encoder scan, one wave32.  lane j <-> hidden unit j.
// ---------------------------------------------------------------------------
__device__ __forceinline__ float mv32(const float* __restrict__ hv,
                                      const float* __restrict__ W, int lane) {
  float acc = 0.0f;
#pragma unroll
  for (int i = 0; i < H; ++i) acc = fmaf(hv[i], W[i * H + lane], acc);
  return acc;
}

__global__ void __launch_bounds__(32)
k2_encoder(const float* __restrict__ xfp, const float* __restrict__ xpp,
           const float* __restrict__ syl_seq,
           const float* __restrict__ Whf0, const float* __restrict__ bhf0_g,
           const float* __restrict__ Whf1, const float* __restrict__ Wxf1,
           const float* __restrict__ bhf1_g, const float* __restrict__ bxf1_g,
           const float* __restrict__ Whp0, const float* __restrict__ bhp0_g,
           const float* __restrict__ Whp1, const float* __restrict__ Wxp1,
           const float* __restrict__ bhp1_g, const float* __restrict__ bxp1_g,
           const float* __restrict__ Whs, const float* __restrict__ Wxs,
           const float* __restrict__ bhs_g, const float* __restrict__ bxs_g,
           const int* __restrict__ frnn_clock, const int* __restrict__ phrnn_clock,
           const int* __restrict__ sample_freq,
           float* __restrict__ ys_sel, int* __restrict__ Kout, int T) {
  __shared__ float sW[8][H * H];           // 8 x 4KB = 32KB
  __shared__ float sh_hf0[H], sh_hf[H], sh_hp0[H], sh_hp[H], sh_hs[H], sh_res[H];

  int lane = threadIdx.x;
  const float* gW[8] = {Whf0, Wxf1, Whf1, Whp0, Wxp1, Whp1, Wxs, Whs};
#pragma unroll
  for (int mtx = 0; mtx < 8; ++mtx)
    for (int i = lane; i < H * H; i += 32) sW[mtx][i] = gW[mtx][i];

  float bhf0 = bhf0_g[lane], bxf1 = bxf1_g[lane], bhf1 = bhf1_g[lane];
  float bhp0 = bhp0_g[lane], bxp1 = bxp1_g[lane], bhp1 = bhp1_g[lane];
  float bxs = bxs_g[lane], bhs = bhs_g[lane];

  float hf0 = 0.f, hf = 0.f, hp0 = 0.f, hp = 0.f, hs = 0.f;
  sh_hf0[lane] = 0.f; sh_hf[lane] = 0.f; sh_hp0[lane] = 0.f;
  sh_hp[lane] = 0.f; sh_hs[lane] = 0.f;
  __syncthreads();

  int k = 0;
  for (int t = 0; t < T; ++t) {
    if (t + 8 < T) {
      __builtin_prefetch(&xfp[(long)(t + 8) * H], 0, 1);
      __builtin_prefetch(&xpp[(long)(t + 8) * H], 0, 1);
      __builtin_prefetch(&syl_seq[(long)(t + 8) * H], 0, 1);
    }
    float xf = xfp[(long)t * H + lane];
    float xp = xpp[(long)t * H + lane];
    float xs = syl_seq[(long)t * H + lane];
    int cf = __builtin_amdgcn_readfirstlane(frnn_clock[t]);
    int cp = __builtin_amdgcn_readfirstlane(phrnn_clock[t]);
    int sf = __builtin_amdgcn_readfirstlane(sample_freq[t]);
    bool gf = (t % cf) == 0;
    bool gp = (t % cp) == 0;

    // frnn0
    float cand = tanhf(xf + mv32(sh_hf0, sW[0], lane) + bhf0);
    hf0 = gf ? cand : hf0;
    __syncthreads(); sh_hf0[lane] = hf0; __syncthreads();
    // frnn1 (gelu applied to post-gate value, per reference)
    float pre = mv32(sh_hf0, sW[1], lane) + bxf1 + mv32(sh_hf, sW[2], lane) + bhf1;
    hf = gelu_exact(gf ? tanhf(pre) : hf);
    __syncthreads(); sh_hf[lane] = hf; __syncthreads();
    // phrnn0
    cand = tanhf(xp + mv32(sh_hp0, sW[3], lane) + bhp0);
    hp0 = gp ? cand : hp0;
    __syncthreads(); sh_hp0[lane] = hp0; __syncthreads();
    // phrnn1
    pre = mv32(sh_hp0, sW[4], lane) + bxp1 + mv32(sh_hp, sW[5], lane) + bhp1;
    hp = gelu_exact(gp ? tanhf(pre) : hp);
    __syncthreads(); sh_hp[lane] = hp; __syncthreads();
    // sylrnn (t argument is 0 in reference -> always candidate), gated by sf
    float res = hf + hp + xs;
    sh_res[lane] = res; __syncthreads();
    float hsc = tanhf(mv32(sh_res, sW[6], lane) + bxs + mv32(sh_hs, sW[7], lane) + bhs);
    float hs_new = gelu_exact(hsc);
    hs = (sf == 1) ? hs_new : hs;
    __syncthreads(); sh_hs[lane] = hs; __syncthreads();

    if (sf == 1) { ys_sel[(long)k * H + lane] = hs; ++k; }
  }
  if (lane == 0) Kout[0] = k;
}

// ---------------------------------------------------------------------------
// K3: parallel decoder front-end over K rows:
//   mean/logvar = ys @ Wm/Wlv + b ; z = mean + eps*exp(0.5*logvar)
//   hs_new = (i%cd==0) ? gelu(tanh(z @ Wxs + bxs + bhs)) : 0
//   hp_new = (i%cd==0) ? gelu(tanh(hs_new @ Wxpd + bxpd + bhpd)) : 0
// 8 WMMAs per 16-row tile; z/hs tiles staged through LDS to rebuild A-frags.
// ---------------------------------------------------------------------------
__global__ void __launch_bounds__(32)
k3_dec_parallel(const float* __restrict__ ys, const float* __restrict__ eps,
                const float* __restrict__ Wm, const float* __restrict__ bm,
                const float* __restrict__ Wlv, const float* __restrict__ blv,
                const float* __restrict__ Wxs, const float* __restrict__ bxs,
                const float* __restrict__ bhs,
                const float* __restrict__ Wxpd, const float* __restrict__ bxpd,
                const float* __restrict__ bhpd,
                const int* __restrict__ dec_clock, const int* __restrict__ Kp,
                float* __restrict__ hs_buf, float* __restrict__ hp_buf,
                float* __restrict__ mean_last, float* __restrict__ logvar_last) {
  __shared__ _Float16 zt[16 * 32];
  __shared__ _Float16 hst[16 * 32];
  int lane = threadIdx.x;
  int K = Kp[0];
  int row0 = blockIdx.x * 16;
  if (row0 >= K) return;
  int mmax = K - row0; if (mmax > 16) mmax = 16;
  int half = lane >> 4, n = lane & 15;

  // --- mean / logvar ---
  v16h A = afrag_f32(ys + (long)row0 * H, H, H, mmax, lane);
  v16h Bm0 = bfrag_f32(Wm, H, 0, H, lane),  Bm1 = bfrag_f32(Wm, H, 16, H, lane);
  v16h Bl0 = bfrag_f32(Wlv, H, 0, H, lane), Bl1 = bfrag_f32(Wlv, H, 16, H, lane);
  v8f cm0 = {}, cm1 = {}, cl0 = {}, cl1 = {};
  cm0 = wmma_f16(A, Bm0, cm0); cm1 = wmma_f16(A, Bm1, cm1);
  cl0 = wmma_f16(A, Bl0, cl0); cl1 = wmma_f16(A, Bl1, cl1);

#pragma unroll
  for (int r = 0; r < 8; ++r) {
    int lrow = r + 8 * half;
    long row = row0 + lrow;
    bool valid = lrow < mmax;
#pragma unroll
    for (int nt = 0; nt < 2; ++nt) {
      int col = nt * 16 + n;
      float m  = (nt ? cm1[r] : cm0[r]) + bm[col];
      float lv = (nt ? cl1[r] : cl0[r]) + blv[col];
      float ev = valid ? eps[row * H + col] : 0.0f;
      float z = m + ev * expf(0.5f * lv);
      if (!valid) z = 0.0f;
      zt[lrow * H + col] = (_Float16)z;
      if (valid && row == (long)K - 1) { mean_last[col] = m; logvar_last[col] = lv; }
    }
  }
  __syncthreads();

  // --- hs_new = gate(gelu(tanh(z @ Wxs + bxs + bhs))) ---
  v16h A2 = afrag_lds(zt, lane);
  v16h Bs0 = bfrag_f32(Wxs, H, 0, H, lane), Bs1 = bfrag_f32(Wxs, H, 16, H, lane);
  v8f cs0 = {}, cs1 = {};
  cs0 = wmma_f16(A2, Bs0, cs0); cs1 = wmma_f16(A2, Bs1, cs1);

#pragma unroll
  for (int r = 0; r < 8; ++r) {
    int lrow = r + 8 * half;
    long row = row0 + lrow;
    bool valid = lrow < mmax;
    int cd = valid ? dec_clock[row] : 1;
    bool on = ((int)row % cd) == 0;
#pragma unroll
    for (int nt = 0; nt < 2; ++nt) {
      int col = nt * 16 + n;
      float acc = (nt ? cs1[r] : cs0[r]);
      float hv = 0.0f;
      if (valid && on) hv = gelu_exact(tanhf(acc + bxs[col] + bhs[col]));
      hst[lrow * H + col] = (_Float16)hv;
      if (valid) hs_buf[row * H + col] = hv;
    }
  }
  __syncthreads();

  // --- hp_new = gate(gelu(tanh(hs_new @ Wxpd + bxpd + bhpd))) ---
  v16h A3 = afrag_lds(hst, lane);
  v16h Bp0 = bfrag_f32(Wxpd, H, 0, H, lane), Bp1 = bfrag_f32(Wxpd, H, 16, H, lane);
  v8f cp0 = {}, cp1 = {};
  cp0 = wmma_f16(A3, Bp0, cp0); cp1 = wmma_f16(A3, Bp1, cp1);

#pragma unroll
  for (int r = 0; r < 8; ++r) {
    int lrow = r + 8 * half;
    long row = row0 + lrow;
    bool valid = lrow < mmax;
    int cd = valid ? dec_clock[row] : 1;
    bool on = ((int)row % cd) == 0;
#pragma unroll
    for (int nt = 0; nt < 2; ++nt) {
      int col = nt * 16 + n;
      float acc = (nt ? cp1[r] : cp0[r]);
      float hv = 0.0f;
      if (valid && on) hv = gelu_exact(tanhf(acc + bxpd[col] + bhpd[col]));
      if (valid) hp_buf[row * H + col] = hv;
    }
  }
}

// ---------------------------------------------------------------------------
// K4: serial decoder tail, one wave32: hp_d last-valid select, hff scalar
// recurrence, hfc 12-wide clocked recurrence, KL from last row.
// ---------------------------------------------------------------------------
__global__ void __launch_bounds__(32)
k4_dec_serial(const float* __restrict__ hp_buf, const int* __restrict__ sample_freq,
              const float* __restrict__ Whf, const float* __restrict__ Wxf,
              const float* __restrict__ bhf, const float* __restrict__ bxf,
              const float* __restrict__ Whc, const float* __restrict__ Wxc,
              const float* __restrict__ bhc, const float* __restrict__ bxc,
              const int* __restrict__ dec_clock_c, const int* __restrict__ Kp,
              const float* __restrict__ mean_last, const float* __restrict__ logvar_last,
              float* __restrict__ out) {
  __shared__ float sWxf[H];          // 32x1
  __shared__ float sWxc[H * 12];     // 32x12
  __shared__ float sWhc[12 * 12];    // 12x12
  __shared__ float sh_hp[H];
  __shared__ float sh_hfc[12];

  int lane = threadIdx.x;
  int K = Kp[0];

  sWxf[lane] = Wxf[lane];
  for (int i = lane; i < H * 12; i += 32) sWxc[i] = Wxc[i];
  for (int i = lane; i < 12 * 12; i += 32) sWhc[i] = Whc[i];
  if (lane < 12) sh_hfc[lane] = 0.0f;
  float whf = Whf[0], bhf0 = bhf[0], bxf0 = bxf[0];
  __syncthreads();

  float hff = 0.0f;
  float hfc = 0.0f;                       // lane<12 meaningful
  float a_f = bxf0;                       // hp_d = 0 initially
  float a_c = (lane < 12) ? bxc[lane] : 0.0f;
  float bhcv = (lane < 12) ? bhc[lane] : 0.0f;

  for (int i = 0; i < K; ++i) {
    if (i + 8 < K) __builtin_prefetch(&hp_buf[(long)(i + 8) * H], 0, 1);
    int sfprev = (i > 0) ? __builtin_amdgcn_readfirstlane(sample_freq[i - 1]) : 0;
    if (sfprev == 1) {                    // hp_d takes this step's hp_new
      sh_hp[lane] = hp_buf[(long)i * H + lane];
      __syncthreads();
      float af = 0.0f;
#pragma unroll
      for (int l = 0; l < H; ++l) af = fmaf(sh_hp[l], sWxf[l], af);
      a_f = af + bxf0;
      if (lane < 12) {
        float ac = 0.0f;
#pragma unroll
        for (int l = 0; l < H; ++l) ac = fmaf(sh_hp[l], sWxc[l * 12 + lane], ac);
        a_c = ac + bxc[lane];
      }
      __syncthreads();
    }
    // hff: clock 1 -> always updates
    hff = tanhf(a_f + whf * hff + bhf0);
    // hfc: clocked
    int cc = __builtin_amdgcn_readfirstlane(dec_clock_c[i]);
    if ((i % cc) == 0) {
      float nv = 0.0f;
      if (lane < 12) {
        float s = a_c + bhcv;
#pragma unroll
        for (int l = 0; l < 12; ++l) s = fmaf(sh_hfc[l], sWhc[l * 12 + lane], s);
        nv = tanhf(s);
      }
      __syncthreads();
      if (lane < 12) { sh_hfc[lane] = nv; hfc = nv; }
      __syncthreads();
    }
  }

  // KL of last decoder row (reference overwrites kl each step)
  float kl = 0.0f;
  if (K > 0) {
    float m = mean_last[lane], lv = logvar_last[lane];
    float term = 1.0f + lv - m * m - expf(lv);
#pragma unroll
    for (int off = 16; off > 0; off >>= 1) term += __shfl_xor(term, off, 32);
    kl = -0.5f * term;
  }
  if (lane == 0) { out[0] = hff; out[13] = kl; }
  if (lane < 12) out[1 + lane] = hfc;
}

// ---------------------------------------------------------------------------
// Host-side launch.  Input order: setup_inputs() insertion order, with the
// nested `params` dict flattened as a JAX pytree (dict keys sorted, tuples in
// order): bn_logvar(W,b), bn_mean(W,b), frnn0..sylrnn each as (Wh,Wx,bh,bx).
// ---------------------------------------------------------------------------
extern "C" void kernel_launch(void* const* d_in, const int* in_sizes, int n_in,
                              void* d_out, int out_size, void* d_ws, size_t ws_size,
                              hipStream_t stream) {
  const float* frnn_seq  = (const float*)d_in[0];   // T x 13
  const float* phrnn_seq = (const float*)d_in[1];   // T x 7
  const float* syl_seq   = (const float*)d_in[2];   // T x 32
  const float* eps       = (const float*)d_in[3];   // T x 32
  // params (pytree order)
  const float* Wlv  = (const float*)d_in[4];
  const float* blv  = (const float*)d_in[5];
  const float* Wm   = (const float*)d_in[6];
  const float* bm   = (const float*)d_in[7];
  const float* Whf0 = (const float*)d_in[8];
  const float* Wxf0 = (const float*)d_in[9];
  const float* bhf0 = (const float*)d_in[10];
  const float* bxf0 = (const float*)d_in[11];
  const float* Whf1 = (const float*)d_in[12];
  const float* Wxf1 = (const float*)d_in[13];
  const float* bhf1 = (const float*)d_in[14];
  const float* bxf1 = (const float*)d_in[15];
  const float* Whc  = (const float*)d_in[16];
  const float* Wxc  = (const float*)d_in[17];
  const float* bhc  = (const float*)d_in[18];
  const float* bxc  = (const float*)d_in[19];
  const float* Whf  = (const float*)d_in[20];
  const float* Wxf  = (const float*)d_in[21];
  const float* bhf  = (const float*)d_in[22];
  const float* bxf  = (const float*)d_in[23];
  const float* Whp0 = (const float*)d_in[24];
  const float* Wxp0 = (const float*)d_in[25];
  const float* bhp0 = (const float*)d_in[26];
  const float* bxp0 = (const float*)d_in[27];
  const float* Whp1 = (const float*)d_in[28];
  const float* Wxp1 = (const float*)d_in[29];
  const float* bhp1 = (const float*)d_in[30];
  const float* bxp1 = (const float*)d_in[31];
  const float* Whpd = (const float*)d_in[32];
  const float* Wxpd = (const float*)d_in[33];
  const float* bhpd = (const float*)d_in[34];
  const float* bxpd = (const float*)d_in[35];
  const float* Whs  = (const float*)d_in[36];
  const float* Wxs  = (const float*)d_in[37];
  const float* bhs  = (const float*)d_in[38];
  const float* bxs  = (const float*)d_in[39];
  const int* frnn_clock  = (const int*)d_in[40];
  const int* phrnn_clock = (const int*)d_in[41];
  const int* sample_freq = (const int*)d_in[42];
  // d_in[43] = syl_clock (unused: reference passes t=0 to sylrnn _cw)
  const int* dec_clock   = (const int*)d_in[44];
  const int* dec_clock_c = (const int*)d_in[45];

  const int T = in_sizes[0] / 13;

  // workspace carve-up (~84 MB for T=131072)
  char* w = (char*)d_ws;
  float* xfp    = (float*)w; w += (size_t)T * H * sizeof(float);
  float* xpp    = (float*)w; w += (size_t)T * H * sizeof(float);
  float* ys_sel = (float*)w; w += (size_t)T * H * sizeof(float);
  float* hs_buf = (float*)w; w += (size_t)T * H * sizeof(float);
  float* hp_buf = (float*)w; w += (size_t)T * H * sizeof(float);
  float* mean_last   = (float*)w; w += H * sizeof(float);
  float* logvar_last = (float*)w; w += H * sizeof(float);
  int* Kp = (int*)w; w += 4 * sizeof(int);
  (void)ws_size; (void)n_in; (void)out_size;

  // K1: input projections (WMMA), 16 rows per wave, 4 waves per block
  int tiles = (T + 15) / 16;
  int blocks = (tiles + 3) / 4;
  k1_inproj<<<blocks, 128, 0, stream>>>(frnn_seq, phrnn_seq, Wxf0, bxf0, Wxp0,
                                        bxp0, xfp, xpp, T);

  // K2: serial encoder scan + compaction (single wave32)
  k2_encoder<<<1, 32, 0, stream>>>(xfp, xpp, syl_seq,
                                   Whf0, bhf0, Whf1, Wxf1, bhf1, bxf1,
                                   Whp0, bhp0, Whp1, Wxp1, bhp1, bxp1,
                                   Whs, Wxs, bhs, bxs,
                                   frnn_clock, phrnn_clock, sample_freq,
                                   ys_sel, Kp, T);

  // K3: decoder parallel front-end (WMMA); grid sized for max K, blocks
  // early-exit on device-side K
  k3_dec_parallel<<<tiles, 32, 0, stream>>>(ys_sel, eps, Wm, bm, Wlv, blv,
                                            Wxs, bxs, bhs, Wxpd, bxpd, bhpd,
                                            dec_clock, Kp, hs_buf, hp_buf,
                                            mean_last, logvar_last);

  // K4: serial decoder tail (single wave32) -> out[14] = {hff, hfc[12], kl}
  k4_dec_serial<<<1, 32, 0, stream>>>(hp_buf, sample_freq,
                                      Whf, Wxf, bhf, bxf,
                                      Whc, Wxc, bhc, bxc,
                                      dec_clock_c, Kp, mean_last, logvar_last,
                                      (float*)d_out);
}
